// Attention_67113158967927
// MI455X (gfx1250) — matrix operand order repaired and checked by reading, not compiled, not run on hardware
//
#include <hip/hip_runtime.h>
#include <math.h>

// ---------------- problem dims ----------------
#define Bdim 64
#define Sdim 512
#define Hdim 1024
#define Adim 1024
#define Edim 2048              // 2*H
#define Mrows (Bdim * Sdim)    // 32768 GEMM rows

typedef __attribute__((ext_vector_type(16))) __bf16        v16bf;
typedef __attribute__((ext_vector_type(8)))  float         v8f;
typedef __attribute__((ext_vector_type(4)))  float         v4f;
typedef __attribute__((ext_vector_type(4)))  unsigned int  v4u;

// round-to-nearest-even fp32 -> bf16, packed pair into one dword
__device__ __forceinline__ unsigned int f2bf_pk(float lo, float hi) {
  union { float f; unsigned int u; } a, b;
  a.f = lo; b.f = hi;
  unsigned int ra = a.u + 0x7FFFu + ((a.u >> 16) & 1u);
  unsigned int rb = b.u + 0x7FFFu + ((b.u >> 16) & 1u);
  return (ra >> 16) | (rb & 0xFFFF0000u);
}

// WGP-scope prefetch (SCOPE field = 0): pulls the line into all cache levels.
// __builtin_prefetch can only reach SE/DEV/SYS scopes, which skip the WGP$.
__device__ __forceinline__ void prefetch_wgp(const void* p) {
  asm volatile("global_prefetch_b8 %0, off" : : "v"(p));
}

// ---------------------------------------------------------------------------
// K0a: repack w_enc (= attn_w rows [H, 3H)) fp32 -> bf16 WMMA B-fragments.
// Fragment (ntile, ktile): 32 lanes x 16 bf16 (1 KB): lane L<16 holds column
// n0+L, K = k0..k0+15 (2 bf16/dword); lane L>=16 holds K = k0+16..k0+31.
// Fragment order: [ntile][ktile] (K-contiguous per wave).
// ---------------------------------------------------------------------------
__global__ void pack_w_enc(const float* __restrict__ attn_w,
                           unsigned int* __restrict__ packB) {
  int id   = blockIdx.x * blockDim.x + threadIdx.x;  // 64*64*32 threads
  int lane = id & 31;
  int kt   = (id >> 5) & 63;
  int nt   = id >> 11;                               // 0..63
  int n     = nt * 16 + (lane & 15);
  int kbase = kt * 32 + ((lane >> 4) << 4);
  const float* src = attn_w + (size_t)(Hdim + kbase) * Adim + n;
  unsigned int* dst = packB + (size_t)(nt * 64 + kt) * 256 + lane * 8;
#pragma unroll
  for (int j = 0; j < 8; ++j) {
    float f0 = src[(2 * j) * Adim];
    float f1 = src[(2 * j + 1) * Adim];
    dst[j] = f2bf_pk(f0, f1);
  }
}

// ---------------------------------------------------------------------------
// K0b: dec_proj = decoder_hidden @ w_dec + attn_b   (64 x 1024, tiny)
// ---------------------------------------------------------------------------
__global__ void dec_proj_kernel(const float* __restrict__ dh,
                                const float* __restrict__ attn_w,
                                const float* __restrict__ attn_b,
                                float* __restrict__ dec_proj) {
  __shared__ float sh[Hdim];
  int b = blockIdx.y;
  int a = blockIdx.x * 128 + threadIdx.x;
  for (int k = threadIdx.x; k < Hdim; k += 128) sh[k] = dh[b * Hdim + k];
  __syncthreads();
  float acc = attn_b[a];
  for (int k = 0; k < Hdim; ++k) acc += sh[k] * attn_w[(size_t)k * Adim + a];
  dec_proj[b * Adim + a] = acc;
}

// ---------------------------------------------------------------------------
// K1: fused  enc_proj GEMM (bf16 WMMA, fp32 accum) + tanh(+dec_proj) . v_w
// Block: 256 threads = 8 waves, owns 64 GEMM rows.
// A tile (64 rows x 32 K): cooperatively loaded fp32, converted to bf16 once
// per block, staged double-buffered in LDS in WMMA A-fragment layout.
// Wave w: N-slice [w*128, w*128+128) -> 8 N-tiles x 4 M-tiles = 32 acc tiles.
// ---------------------------------------------------------------------------
__global__ __launch_bounds__(256, 1)
void attn_energy_kernel(const float* __restrict__ enc,
                        const unsigned int* __restrict__ packB,
                        const float* __restrict__ dec_proj,
                        const float* __restrict__ v_w,
                        float* __restrict__ attn) {
  __shared__ __align__(16) unsigned int abuf[2][1024];  // 2 x 4KB A fragments
  __shared__ float red[8][64];

  const int tid   = threadIdx.x;
  const int lane  = tid & 31;
  const int wave  = tid >> 5;
  const int m0    = blockIdx.x * 64;
  const int b     = m0 >> 9;          // 512 rows per batch, 64 | 512
  const int hih   = lane >> 4;        // half-wave select
  const int lanem = lane & 15;

  // converter role: this thread owns (row crow, 8-float K-chunk cchk).
  // A-fragment layout: lanes 0-15 -> K{0..15}, lanes 16-31 -> K{16..31}
  const int crow  = tid >> 2;                         // 0..63
  const int cchk  = tid & 3;                          // 0..3 (K octets)
  const int cmt   = crow >> 4;
  const int clane = ((cchk & 2) ? 16 : 0) + (crow & 15);
  const int cdst  = cmt * 256 + clane * 8 + (cchk & 1) * 4;  // dword index
  const float* csrc = enc + (size_t)(m0 + crow) * Edim + cchk * 8;

  const v8f vzero = {0.f, 0.f, 0.f, 0.f, 0.f, 0.f, 0.f, 0.f};
  v8f acc[4][8];
#pragma unroll
  for (int mt = 0; mt < 4; ++mt)
#pragma unroll
    for (int nt = 0; nt < 8; ++nt) acc[mt][nt] = vzero;

  const unsigned int* bbase = packB + (size_t)(wave * 8) * 64 * 256 + lane * 8;

  // stage kt=0 into abuf[0]; prefetch kt=1 into registers
  v4f g0 = *(const v4f*)(csrc);
  v4f g1 = *(const v4f*)(csrc + 4);
  {
    v4u u;
    u.x = f2bf_pk(g0.x, g0.y);  u.y = f2bf_pk(g0.z, g0.w);
    u.z = f2bf_pk(g1.x, g1.y);  u.w = f2bf_pk(g1.z, g1.w);
    *(v4u*)&abuf[0][cdst] = u;
  }
  g0 = *(const v4f*)(csrc + 32);
  g1 = *(const v4f*)(csrc + 36);
  __syncthreads();

  int cur = 0;
  for (int kt = 0; kt < 64; ++kt) {
    // 1) read the 4 A fragments of this K-step from LDS
    v16bf afrag[4];
#pragma unroll
    for (int mt = 0; mt < 4; ++mt) {
      union { v16bf v; v4u q[2]; } A;
      const unsigned int* ap = &abuf[cur][mt * 256 + lane * 8];
      A.q[0] = *(const v4u*)(ap);
      A.q[1] = *(const v4u*)(ap + 4);
      afrag[mt] = A.v;
    }

    // 2) stage NEXT K tile into the other LDS buffer now, so the ds_store
    //    latency overlaps with the WMMA block below; prefetch the A tile
    //    after that into registers.
    if (kt < 63) {
      v4u u;
      u.x = f2bf_pk(g0.x, g0.y);  u.y = f2bf_pk(g0.z, g0.w);
      u.z = f2bf_pk(g1.x, g1.y);  u.w = f2bf_pk(g1.z, g1.w);
      *(v4u*)&abuf[cur ^ 1][cdst] = u;
      if (kt < 62) {
        g0 = *(const v4f*)(csrc + (kt + 2) * 32);
        g1 = *(const v4f*)(csrc + (kt + 2) * 32 + 4);
      }
    }

    // 3) load ALL 8 B fragments for this K-step up front (deep pipelining),
    //    and prefetch next K-step's B stream into the WGP cache.
    v16bf bfrag[8];
#pragma unroll
    for (int nt = 0; nt < 8; ++nt) {
      const unsigned int* bp = bbase + ((size_t)nt * 64 + kt) * 256;
      union { v16bf v; v4u q[2]; } Bf;
      Bf.q[0] = *(const v4u*)(bp);
      Bf.q[1] = *(const v4u*)(bp + 4);
      bfrag[nt] = Bf.v;
      if (kt < 63) prefetch_wgp(bp + 256);
    }

    // 4) 32 WMMAs
#pragma unroll
    for (int nt = 0; nt < 8; ++nt)
#pragma unroll
      for (int mt = 0; mt < 4; ++mt)
        acc[mt][nt] = __builtin_amdgcn_wmma_f32_16x16x32_bf16(
            false, afrag[mt], false, bfrag[nt], (short)0, acc[mt][nt], false, false);

    __syncthreads();
    cur ^= 1;
  }

  // epilogue: energy = tanh(acc + dec_proj[b, n]); partial = energy . v_w
  float dp[8], vw[8];
#pragma unroll
  for (int nt = 0; nt < 8; ++nt) {
    int n = wave * 128 + nt * 16 + lanem;
    dp[nt] = dec_proj[b * Adim + n];
    vw[nt] = v_w[n];
  }
#pragma unroll
  for (int mt = 0; mt < 4; ++mt) {
#pragma unroll
    for (int r = 0; r < 8; ++r) {
      float s = 0.f;
#pragma unroll
      for (int nt = 0; nt < 8; ++nt)
        s += tanhf(acc[mt][nt][r] + dp[nt]) * vw[nt];
      // butterfly reduce across the 16 lanes that share this row
      s += __shfl_xor(s, 1, 32);
      s += __shfl_xor(s, 2, 32);
      s += __shfl_xor(s, 4, 32);
      s += __shfl_xor(s, 8, 32);
      if (lanem == 0) red[wave][mt * 16 + hih * 8 + r] = s;
    }
  }
  __syncthreads();
  if (threadIdx.x < 64) {
    float t = 0.f;
#pragma unroll
    for (int w = 0; w < 8; ++w) t += red[w][threadIdx.x];
    attn[m0 + threadIdx.x] = t;
  }
}

// ---------------------------------------------------------------------------
// K2: masked softmax over S per batch row
// ---------------------------------------------------------------------------
__global__ void softmax_kernel(const float* __restrict__ attn,
                               const int* __restrict__ mask,
                               float* __restrict__ out_w) {
  __shared__ float sh[Sdim];
  int b = blockIdx.x, t = threadIdx.x;
  float x = attn[b * Sdim + t];
  if (mask[b * Sdim + t] == 0) x = -__builtin_inff();
  sh[t] = x;
  __syncthreads();
  for (int off = Sdim / 2; off > 0; off >>= 1) {
    if (t < off) sh[t] = fmaxf(sh[t], sh[t + off]);
    __syncthreads();
  }
  float m = sh[0];
  __syncthreads();
  float e = expf(x - m);
  sh[t] = e;
  __syncthreads();
  for (int off = Sdim / 2; off > 0; off >>= 1) {
    if (t < off) sh[t] += sh[t + off];
    __syncthreads();
  }
  out_w[b * Sdim + t] = e / sh[0];
}

// ---------------------------------------------------------------------------
// K3: context = attention_weights @ encoder_outputs   (HBM bound, 256 MB)
// ---------------------------------------------------------------------------
__global__ void context_kernel(const float* __restrict__ enc,
                               const float* __restrict__ w,
                               float* __restrict__ ctx) {
  __shared__ float sw[Sdim];
  int b = blockIdx.y;
  int e = blockIdx.x * 256 + threadIdx.x;
  for (int s = threadIdx.x; s < Sdim; s += 256) sw[s] = w[b * Sdim + s];
  __syncthreads();
  float acc = 0.f;
  for (int s = 0; s < Sdim; ++s)
    acc += sw[s] * enc[(size_t)(b * Sdim + s) * Edim + e];
  ctx[(size_t)b * Edim + e] = acc;
}

// ---------------------------------------------------------------------------
extern "C" void kernel_launch(void* const* d_in, const int* in_sizes, int n_in,
                              void* d_out, int out_size, void* d_ws, size_t ws_size,
                              hipStream_t stream) {
  const float* dh     = (const float*)d_in[0];   // (64, 1024)
  const float* enc    = (const float*)d_in[1];   // (64, 512, 2048)
  const int*   mask   = (const int*)d_in[2];     // (64, 512)
  const float* attn_w = (const float*)d_in[3];   // (3072, 1024)
  const float* attn_b = (const float*)d_in[4];   // (1024,)
  const float* v_w    = (const float*)d_in[5];   // (1024,)

  float* ctx      = (float*)d_out;               // (64, 2048)
  float* attw_out = (float*)d_out + Bdim * Edim; // (64, 512)

  unsigned int* packB    = (unsigned int*)d_ws;                          // 4 MB
  float*        dec_proj = (float*)((char*)d_ws + ((size_t)4 << 20));    // 256 KB
  float*        attn     = dec_proj + Bdim * Adim;                       // 128 KB

  pack_w_enc      <<<512, 256, 0, stream>>>(attn_w, packB);
  dec_proj_kernel <<<dim3(8, 64), 128, 0, stream>>>(dh, attn_w, attn_b, dec_proj);
  attn_energy_kernel<<<Mrows / 64, 256, 0, stream>>>(enc, packB, dec_proj, v_w, attn);
  softmax_kernel  <<<Bdim, Sdim, 0, stream>>>(attn, mask, attw_out);
  context_kernel  <<<dim3(Edim / 256, Bdim), 256, 0, stream>>>(enc, attw_out, ctx);
}
